// LLMBlock_15015205667496
// MI455X (gfx1250) — matrix operand order repaired and checked by reading, not compiled, not run on hardware
//
#include <hip/hip_runtime.h>

typedef __bf16 bf16_t;
typedef __attribute__((ext_vector_type(16))) __bf16 v16bf;
typedef __attribute__((ext_vector_type(8)))  __bf16 v8bf;
typedef __attribute__((ext_vector_type(8)))  float  v8f;

constexpr int CB  = 2;
constexpr int CS  = 2048;
constexpr int CD  = 2048;
constexpr int CH  = 16;
constexpr int CKVH = 8;
constexpr int CHD = 128;
constexpr int CFF = 5504;
constexpr int CM  = CB * CS;               // 4096 tokens
constexpr float QK_SCALE = 0.08838834764831843f;  // 1/sqrt(128)
constexpr float RMS_EPS  = 1e-5f;

// ---------------------------------------------------------------------------
// WMMA fragment loaders (CDNA5 16x16x32 bf16 layouts, wave32)
// A (16x32, MxK): lanes 0-15 row M=lane, K={0..7,16..23}; lanes 16-31 row
//                 M=lane-16, K={8..15,24..31}. 2 bf16 per VGPR.
// B (32x16, KxN): lanes 0-15 col N=lane, K={0..7,16..23}; lanes 16-31 col
//                 N=lane-16, K={8..15,24..31}.
// ---------------------------------------------------------------------------
__device__ __forceinline__ v16bf load_frag_a(const bf16_t* p, int hf) {
  v16bf f;
  ((v8bf*)&f)[0] = *(const v8bf*)(p + (hf ? 8 : 0));
  ((v8bf*)&f)[1] = *(const v8bf*)(p + (hf ? 24 : 16));
  return f;
}
__device__ __forceinline__ v16bf load_frag_b(const bf16_t* p, int hf) {
  v16bf f;
  const bf16_t* q = p + (hf ? 8 : 0);
  ((v8bf*)&f)[0] = *(const v8bf*)(q);
  ((v8bf*)&f)[1] = *(const v8bf*)(q + 16);
  return f;
}

// ---------------------------------------------------------------------------
// fp32 -> bf16 elementwise convert (weights)
// ---------------------------------------------------------------------------
__global__ __launch_bounds__(256) void cvt_bf16_kernel(const float* __restrict__ in,
                                                       bf16_t* __restrict__ out, int n) {
  int i = blockIdx.x * blockDim.x + threadIdx.x;
  if (i < n) out[i] = (__bf16)in[i];
}

// ---------------------------------------------------------------------------
// RMSNorm: per-row (D=2048), fp32 in, bf16 out (GEMM A operand)
// ---------------------------------------------------------------------------
__global__ __launch_bounds__(256) void rmsnorm_kernel(const float* __restrict__ in,
                                                      const float* __restrict__ g,
                                                      bf16_t* __restrict__ out) {
  __shared__ float red[256];
  const int row = blockIdx.x;
  const float* x = in + (size_t)row * CD;
  float ss = 0.f;
  for (int i = threadIdx.x; i < CD; i += 256) { float v = x[i]; ss += v * v; }
  red[threadIdx.x] = ss;
  __syncthreads();
  for (int s = 128; s > 0; s >>= 1) {
    if (threadIdx.x < s) red[threadIdx.x] += red[threadIdx.x + s];
    __syncthreads();
  }
  const float inv = rsqrtf(red[0] / (float)CD + RMS_EPS);
  bf16_t* o = out + (size_t)row * CD;
  for (int i = threadIdx.x; i < CD; i += 256) o[i] = (__bf16)(x[i] * inv * g[i]);
}

// ---------------------------------------------------------------------------
// Generic GEMM: C[M,N] = A[M,K] @ W[N,K]^T (+ resid), bf16 inputs, fp32 out.
// Block = 128 threads = 4 waves; each wave owns a 32x64 register tile
// (2 A frags x 4 B frags = 8 WMMA per K-step; B reused across both row
// strips, A reused across 4 column tiles -> ~16 flop/byte from cache).
// Requires K%32==0, N%64==0, M%32==0.
// ---------------------------------------------------------------------------
__global__ __launch_bounds__(128) void gemm_bf16_kernel(const bf16_t* __restrict__ A,
                                                        const bf16_t* __restrict__ W,
                                                        const float* resid,
                                                        float* C,
                                                        int Mm, int Nn, int Kk) {
  const int lane = threadIdx.x & 31;
  const int wave = threadIdx.x >> 5;
  const int hf   = lane >> 4;
  const int l16  = lane & 15;
  const int m0   = blockIdx.y * 32;
  const int n0   = (blockIdx.x * 4 + wave) * 64;
  if (n0 >= Nn) return;

  const bf16_t* arow0 = A + (size_t)(m0 + l16) * Kk;
  const bf16_t* arow1 = A + (size_t)(m0 + 16 + l16) * Kk;
  const bf16_t* wrow[4];
#pragma unroll
  for (int t = 0; t < 4; ++t) wrow[t] = W + (size_t)(n0 + 16 * t + l16) * Kk;

  v8f zero = {};
  v8f acc0[4], acc1[4];
#pragma unroll
  for (int t = 0; t < 4; ++t) { acc0[t] = zero; acc1[t] = zero; }

  for (int k = 0; k < Kk; k += 32) {
    __builtin_prefetch(arow0 + k + 512, 0, 3);
    __builtin_prefetch(wrow[1] + k + 512, 0, 3);
    __builtin_prefetch(wrow[3] + k + 512, 0, 3);
    v16bf a0 = load_frag_a(arow0 + k, hf);
    v16bf a1 = load_frag_a(arow1 + k, hf);
#pragma unroll
    for (int t = 0; t < 4; ++t) {
      v16bf b = load_frag_b(wrow[t] + k, hf);
      acc0[t] = __builtin_amdgcn_wmma_f32_16x16x32_bf16(false, a0, false, b,
                                                        (short)0, acc0[t], false, false);
      acc1[t] = __builtin_amdgcn_wmma_f32_16x16x32_bf16(false, a1, false, b,
                                                        (short)0, acc1[t], false, false);
    }
  }

#pragma unroll
  for (int t = 0; t < 4; ++t) {
    const int n = n0 + 16 * t + l16;
#pragma unroll
    for (int r = 0; r < 8; ++r) {
      size_t idx0 = (size_t)(m0 + r + 8 * hf) * Nn + n;
      size_t idx1 = (size_t)(m0 + 16 + r + 8 * hf) * Nn + n;
      float v0 = acc0[t][r];
      float v1 = acc1[t][r];
      if (resid) { v0 += resid[idx0]; v1 += resid[idx1]; }
      C[idx0] = v0;
      C[idx1] = v1;
    }
  }
}

// ---------------------------------------------------------------------------
// RoPE + layout change. q fp32 [B,S,H,HD] -> bf16 [B,H,S,HD]
// ---------------------------------------------------------------------------
__global__ __launch_bounds__(256) void rope_q_kernel(const float* __restrict__ q,
                                                     const float* __restrict__ cosb,
                                                     const float* __restrict__ sinb,
                                                     bf16_t* __restrict__ out) {
  const int token = blockIdx.x;
  const int b = token / CS, s = token % CS;
  const float* qr = q + (size_t)token * (CH * CHD);
  for (int i = threadIdx.x; i < CH * CHD; i += 256) {
    const int h = i >> 7, d = i & 127;
    const float v = qr[i];
    const float rot = (d < 64) ? -qr[i + 64] : qr[i - 64];
    const float c = cosb[s * CHD + d], sn = sinb[s * CHD + d];
    out[(((size_t)b * CH + h) * CS + s) * CHD + d] = (__bf16)(v * c + rot * sn);
  }
}

__global__ __launch_bounds__(256) void rope_k_kernel(const float* __restrict__ k,
                                                     const float* __restrict__ cosb,
                                                     const float* __restrict__ sinb,
                                                     bf16_t* __restrict__ out) {
  const int token = blockIdx.x;
  const int b = token / CS, s = token % CS;
  const float* kr = k + (size_t)token * (CKVH * CHD);
  for (int i = threadIdx.x; i < CKVH * CHD; i += 256) {
    const int h = i >> 7, d = i & 127;
    const float v = kr[i];
    const float rot = (d < 64) ? -kr[i + 64] : kr[i - 64];
    const float c = cosb[s * CHD + d], sn = sinb[s * CHD + d];
    out[(((size_t)b * CKVH + h) * CS + s) * CHD + d] = (__bf16)(v * c + rot * sn);
  }
}

// v fp32 [B,S,KVH,HD] -> bf16 transposed [B,KVH,HD,S] (contiguous over keys)
__global__ __launch_bounds__(256) void vtrans_kernel(const float* __restrict__ v,
                                                     bf16_t* __restrict__ out) {
  const int token = blockIdx.x;
  const int b = token / CS, s = token % CS;
  const float* vr = v + (size_t)token * (CKVH * CHD);
  for (int i = threadIdx.x; i < CKVH * CHD; i += 256) {
    const int kh = i >> 7, d = i & 127;
    out[(((size_t)b * CKVH + kh) * CHD + d) * CS + s] = (__bf16)vr[i];
  }
}

// ---------------------------------------------------------------------------
// Flash attention, causal, GQA (REP=2). One wave per 16-row Q tile.
// Q [B,H,S,HD] bf16, K [B,KVH,S,HD] bf16, Vt [B,KVH,HD,S] bf16,
// O [B,S,H*HD] bf16. 32 keys per step: 8 QK WMMA + online softmax + 8 PV WMMA.
// ---------------------------------------------------------------------------
__global__ __launch_bounds__(128) void attn_kernel(const bf16_t* __restrict__ Q,
                                                   const bf16_t* __restrict__ K,
                                                   const bf16_t* __restrict__ Vt,
                                                   bf16_t* __restrict__ O) {
  __shared__ float pt[4][16][33];   // per-wave P staging tile (16 q x 32 k)
  const int lane = threadIdx.x & 31;
  const int wave = threadIdx.x >> 5;
  const int hf   = lane >> 4;
  const int l16  = lane & 15;
  const int b = blockIdx.z, h = blockIdx.y;
  const int q0 = (blockIdx.x * 4 + wave) * 16;
  const int kh = h >> 1;   // REP = 2

  const bf16_t* qbase = Q + (((size_t)b * CH + h) * CS + q0) * CHD;
  const bf16_t* kbase = K + ((size_t)b * CKVH + kh) * (size_t)CS * CHD;
  const bf16_t* vbase = Vt + ((size_t)b * CKVH + kh) * (size_t)CHD * CS;

  v16bf aq[4];
#pragma unroll
  for (int c = 0; c < 4; ++c)
    aq[c] = load_frag_a(qbase + (size_t)l16 * CHD + 32 * c, hf);

  v8f zero = {};
  v8f acc[8];
#pragma unroll
  for (int d = 0; d < 8; ++d) acc[d] = zero;
  float mrow[8], lrow[8];
#pragma unroll
  for (int r = 0; r < 8; ++r) { mrow[r] = -INFINITY; lrow[r] = 0.f; }

  const int kend = q0 + 16;   // exclusive causal bound for this tile
  for (int k0 = 0; k0 < kend; k0 += 32) {
    const bool have2 = (k0 + 16) < kend;
    v8f s0 = zero, s1 = zero;
#pragma unroll
    for (int c = 0; c < 4; ++c) {
      v16bf bk = load_frag_b(kbase + (size_t)(k0 + l16) * CHD + 32 * c, hf);
      s0 = __builtin_amdgcn_wmma_f32_16x16x32_bf16(false, aq[c], false, bk,
                                                   (short)0, s0, false, false);
    }
    if (have2) {
#pragma unroll
      for (int c = 0; c < 4; ++c) {
        v16bf bk = load_frag_b(kbase + (size_t)(k0 + 16 + l16) * CHD + 32 * c, hf);
        s1 = __builtin_amdgcn_wmma_f32_16x16x32_bf16(false, aq[c], false, bk,
                                                     (short)0, s1, false, false);
      }
    }

#pragma unroll
    for (int r = 0; r < 8; ++r) {
      const int qrow = q0 + r + 8 * hf;
      float a0 = s0[r] * QK_SCALE;
      if (k0 + l16 > qrow) a0 = -INFINITY;
      float a1 = have2 ? (s1[r] * QK_SCALE) : -INFINITY;
      if (k0 + 16 + l16 > qrow) a1 = -INFINITY;

      float t = fmaxf(a0, a1);
#pragma unroll
      for (int o = 1; o < 16; o <<= 1) t = fmaxf(t, __shfl_xor(t, o));
      const float mn = fmaxf(mrow[r], t);
      const float pscale = __expf(mrow[r] - mn);
      const float p0 = __expf(a0 - mn);
      const float p1 = __expf(a1 - mn);
      float rs = p0 + p1;
#pragma unroll
      for (int o = 1; o < 16; o <<= 1) rs += __shfl_xor(rs, o);
      lrow[r] = lrow[r] * pscale + rs;
      mrow[r] = mn;
      pt[wave][r + 8 * hf][l16]      = p0;
      pt[wave][r + 8 * hf][16 + l16] = p1;
#pragma unroll
      for (int d = 0; d < 8; ++d) acc[d][r] *= pscale;
    }

    // Re-layout P into an A fragment (LDS round-trip, wave-private tile).
    v16bf pa;
    {
      const float* prow = pt[wave][l16];
      const int o0 = hf ? 8 : 0, o1 = hf ? 24 : 16;
#pragma unroll
      for (int j = 0; j < 8; ++j) {
        pa[j]     = (__bf16)prow[o0 + j];
        pa[8 + j] = (__bf16)prow[o1 + j];
      }
    }
#pragma unroll
    for (int d = 0; d < 8; ++d) {
      v16bf bv = load_frag_b(vbase + (size_t)(d * 16 + l16) * CS + k0, hf);
      acc[d] = __builtin_amdgcn_wmma_f32_16x16x32_bf16(false, pa, false, bv,
                                                       (short)0, acc[d], false, false);
    }
  }

#pragma unroll
  for (int r = 0; r < 8; ++r) {
    const float inv = 1.0f / lrow[r];
    const int srow = q0 + r + 8 * hf;
    const size_t orow = ((size_t)b * CS + srow) * (CH * CHD) + (size_t)h * CHD;
#pragma unroll
    for (int d = 0; d < 8; ++d)
      O[orow + d * 16 + l16] = (__bf16)(acc[d][r] * inv);
  }
}

// ---------------------------------------------------------------------------
// SwiGLU: act = silu(gate) * up, bf16 out
// ---------------------------------------------------------------------------
__global__ __launch_bounds__(256) void swiglu_kernel(const float* __restrict__ g,
                                                     const float* __restrict__ u,
                                                     bf16_t* __restrict__ out, int n) {
  int i = blockIdx.x * blockDim.x + threadIdx.x;
  if (i < n) {
    const float x = g[i];
    const float sig = 1.f / (1.f + __expf(-x));
    out[i] = (__bf16)(x * sig * u[i]);
  }
}

// ---------------------------------------------------------------------------
// Host orchestration
// ---------------------------------------------------------------------------
extern "C" void kernel_launch(void* const* d_in, const int* in_sizes, int n_in,
                              void* d_out, int out_size, void* d_ws, size_t ws_size,
                              hipStream_t stream) {
  (void)in_sizes; (void)n_in; (void)out_size; (void)ws_size;
  const float* x    = (const float*)d_in[0];
  const float* cosb = (const float*)d_in[1];
  const float* sinb = (const float*)d_in[2];
  const float* wq   = (const float*)d_in[3];
  const float* wk   = (const float*)d_in[4];
  const float* wv   = (const float*)d_in[5];
  const float* wo   = (const float*)d_in[6];
  const float* wg   = (const float*)d_in[7];
  const float* wu   = (const float*)d_in[8];
  const float* wd   = (const float*)d_in[9];
  const float* g1   = (const float*)d_in[10];
  const float* g2   = (const float*)d_in[11];
  float* out = (float*)d_out;

  char* ws = (char*)d_ws;
  size_t off = 0;
  auto alloc = [&](size_t bytes) {
    size_t r = off;
    off += (bytes + 255) & ~(size_t)255;
    return r;
  };

  // persistent bf16 weights
  bf16_t* wq_bf = (bf16_t*)(ws + alloc((size_t)CD * CD * 2));
  bf16_t* wk_bf = (bf16_t*)(ws + alloc((size_t)CKVH * CHD * CD * 2));
  bf16_t* wv_bf = (bf16_t*)(ws + alloc((size_t)CKVH * CHD * CD * 2));
  bf16_t* wo_bf = (bf16_t*)(ws + alloc((size_t)CD * CD * 2));
  bf16_t* wg_bf = (bf16_t*)(ws + alloc((size_t)CFF * CD * 2));
  bf16_t* wu_bf = (bf16_t*)(ws + alloc((size_t)CFF * CD * 2));
  bf16_t* wd_bf = (bf16_t*)(ws + alloc((size_t)CD * CFF * 2));
  // normed activations (reused for both RMSNorms)
  bf16_t* h_bf  = (bf16_t*)(ws + alloc((size_t)CM * CD * 2));
  // region A: q/k/v fp32 early, gate fp32 late (lifetimes disjoint)
  char* regA = ws + alloc((size_t)CM * CFF * 4);
  float* qf = (float*)regA;
  float* kf = (float*)(regA + (size_t)CM * CD * 4);
  float* vf = (float*)(regA + (size_t)CM * CD * 4 + (size_t)CM * CKVH * CHD * 4);
  float* gatef = (float*)regA;
  // region B: rotated/transposed bf16 + attn out early, up fp32 late
  char* regB = ws + alloc((size_t)CM * CFF * 4);
  bf16_t* qbf = (bf16_t*)regB;
  bf16_t* kbf = (bf16_t*)(regB + (size_t)CM * CD * 2);
  bf16_t* vtb = (bf16_t*)(regB + (size_t)CM * CD * 2 + (size_t)CM * CKVH * CHD * 2);
  bf16_t* obf = (bf16_t*)(regB + (size_t)CM * CD * 2 + 2 * (size_t)CM * CKVH * CHD * 2 + 4096);
  float* upf = (float*)regB;
  // SwiGLU activation
  bf16_t* act = (bf16_t*)(ws + alloc((size_t)CM * CFF * 2 + 4096));

  auto cdiv = [](long a, long b) { return (int)((a + b - 1) / b); };
  auto gemm = [&](const bf16_t* A, const bf16_t* W, const float* resid, float* C,
                  int Mm, int Nn, int Kk) {
    dim3 grid(cdiv(Nn / 64, 4), Mm / 32);
    gemm_bf16_kernel<<<grid, 128, 0, stream>>>(A, W, resid, C, Mm, Nn, Kk);
  };

  // 1. weight conversion fp32 -> bf16
  cvt_bf16_kernel<<<cdiv((long)CD * CD, 256), 256, 0, stream>>>(wq, wq_bf, CD * CD);
  cvt_bf16_kernel<<<cdiv((long)CKVH * CHD * CD, 256), 256, 0, stream>>>(wk, wk_bf, CKVH * CHD * CD);
  cvt_bf16_kernel<<<cdiv((long)CKVH * CHD * CD, 256), 256, 0, stream>>>(wv, wv_bf, CKVH * CHD * CD);
  cvt_bf16_kernel<<<cdiv((long)CD * CD, 256), 256, 0, stream>>>(wo, wo_bf, CD * CD);
  cvt_bf16_kernel<<<cdiv((long)CFF * CD, 256), 256, 0, stream>>>(wg, wg_bf, CFF * CD);
  cvt_bf16_kernel<<<cdiv((long)CFF * CD, 256), 256, 0, stream>>>(wu, wu_bf, CFF * CD);
  cvt_bf16_kernel<<<cdiv((long)CD * CFF, 256), 256, 0, stream>>>(wd, wd_bf, CD * CFF);

  // 2. pre-attention RMSNorm
  rmsnorm_kernel<<<CM, 256, 0, stream>>>(x, g1, h_bf);

  // 3. QKV projections
  gemm(h_bf, wq_bf, nullptr, qf, CM, CH * CHD, CD);
  gemm(h_bf, wk_bf, nullptr, kf, CM, CKVH * CHD, CD);
  gemm(h_bf, wv_bf, nullptr, vf, CM, CKVH * CHD, CD);

  // 4. RoPE + layout transforms
  rope_q_kernel<<<CM, 256, 0, stream>>>(qf, cosb, sinb, qbf);
  rope_k_kernel<<<CM, 256, 0, stream>>>(kf, cosb, sinb, kbf);
  vtrans_kernel<<<CM, 256, 0, stream>>>(vf, vtb);

  // 5. flash attention
  attn_kernel<<<dim3(CS / 64, CH, CB), 128, 0, stream>>>(qbf, kbf, vtb, obf);

  // 6. output projection + residual  (d_out = x + attn @ wo^T)
  gemm(obf, wo_bf, x, out, CM, CD, CD);

  // 7. post-attention RMSNorm
  rmsnorm_kernel<<<CM, 256, 0, stream>>>(out, g2, h_bf);

  // 8. gate / up projections
  gemm(h_bf, wg_bf, nullptr, gatef, CM, CFF, CD);
  gemm(h_bf, wu_bf, nullptr, upf, CM, CFF, CD);

  // 9. SwiGLU
  swiglu_kernel<<<cdiv((long)CM * CFF, 256), 256, 0, stream>>>(gatef, upf, act, CM * CFF);

  // 10. down projection + residual (in-place on d_out; each element is
  //     read-then-written exactly once by its owning lane)
  gemm(act, wd_bf, out, out, CM, CD, CFF);
}
